// Multihead_Attention_51281909514357
// MI455X (gfx1250) — matrix operand order repaired and checked by reading, not compiled, not run on hardware
//
#include <hip/hip_runtime.h>

typedef __attribute__((ext_vector_type(16))) _Float16 v16h;
typedef __attribute__((ext_vector_type(8)))  _Float16 v8h;
typedef __attribute__((ext_vector_type(8)))  float    v8f;

#define D_MODEL 512
#define S_LEN   2048
#define N_HEADS 8
#define HEAD_D  64
#define N_TOK   (4 * S_LEN)   // 8192

static __device__ __forceinline__ v8f wmma_f16(v16h a, v16h b, v8f c) {
    return __builtin_amdgcn_wmma_f32_16x16x32_f16(
        /*neg_a=*/false, a, /*neg_b=*/false, b,
        /*c_mod=*/(short)0, c, /*reuse_a=*/false, /*reuse_b=*/false);
}

static __device__ __forceinline__ v16h cat16(v8h lo, v8h hi) {
    v16h r;
#pragma unroll
    for (int j = 0; j < 8; ++j) { r[j] = lo[j]; r[8 + j] = hi[j]; }
    return r;
}

// ---------------------------------------------------------------------------
// Kernel 1: convert the four 512x512 fp32 weight matrices to f16
// ---------------------------------------------------------------------------
__global__ void cvt_weights(const float* __restrict__ w0, const float* __restrict__ w1,
                            const float* __restrict__ w2, const float* __restrict__ w3,
                            _Float16* __restrict__ o0, _Float16* __restrict__ o1,
                            _Float16* __restrict__ o2, _Float16* __restrict__ o3) {
    int i = blockIdx.x * 256 + threadIdx.x;   // 0 .. 262143
    o0[i] = (_Float16)w0[i];
    o1[i] = (_Float16)w1[i];
    o2[i] = (_Float16)w2[i];
    o3[i] = (_Float16)w3[i];
}

// ---------------------------------------------------------------------------
// Kernel 2: LayerNorm (eps=1e-5), output f16 row-major [8192, 512]
// one block (256 threads) per token
// ---------------------------------------------------------------------------
__global__ void layernorm_f16(const float* __restrict__ x,
                              const float* __restrict__ gamma,
                              const float* __restrict__ beta,
                              _Float16* __restrict__ xn) {
    const int t   = blockIdx.x;
    const int tid = threadIdx.x;
    const float* row = x + (size_t)t * D_MODEL;
    float v0 = row[tid];
    float v1 = row[tid + 256];

    __shared__ float red[256];
    red[tid] = v0 + v1;
    __syncthreads();
#pragma unroll
    for (int s = 128; s > 0; s >>= 1) {
        if (tid < s) red[tid] += red[tid + s];
        __syncthreads();
    }
    const float mu = red[0] * (1.0f / D_MODEL);
    __syncthreads();

    float d0 = v0 - mu, d1 = v1 - mu;
    red[tid] = d0 * d0 + d1 * d1;
    __syncthreads();
#pragma unroll
    for (int s = 128; s > 0; s >>= 1) {
        if (tid < s) red[tid] += red[tid + s];
        __syncthreads();
    }
    const float inv = rsqrtf(red[0] * (1.0f / D_MODEL) + 1e-5f);

    _Float16* o = xn + (size_t)t * D_MODEL;
    o[tid]       = (_Float16)(d0 * inv * gamma[tid]       + beta[tid]);
    o[tid + 256] = (_Float16)(d1 * inv * gamma[tid + 256] + beta[tid + 256]);
}

// ---------------------------------------------------------------------------
// Kernel 3/5: GEMM  out[m,n] = sum_k A[m,k] * W[n,k] + bias[n]
// A: f16 [8192,512] row-major.  W: f16 [512,512] row-major (rows are columns
// of W^T so B-fragments are contiguous loads).
// Block = 256 threads = 8 waves (2 in M x 4 in N); wave tile = 16x64.
// mode 0: f16 store, layout [B,H,S,Dh]        (Q and K)
// mode 1: f16 store, layout [B,H,Dh,S]        (V transposed)
// mode 2: f32 store + row-major [8192,512]    (final projection)
// ---------------------------------------------------------------------------
__global__ void gemm512(const _Float16* __restrict__ A,
                        const _Float16* __restrict__ W,
                        const float* __restrict__ bias,
                        int mode,
                        _Float16* __restrict__ outH,
                        float* __restrict__ outF) {
    const int w    = threadIdx.x >> 5;       // 0..7
    const int lane = threadIdx.x & 31;
    const int ln   = lane & 15;
    const int hi   = lane >> 4;
    const int wm   = w & 1;                  // 2 waves in M
    const int wn   = w >> 1;                 // 4 waves in N
    const int mTile = blockIdx.x * 32 + wm * 16;
    const int nTile = blockIdx.y * 256 + wn * 64;

    v8f acc[4];
#pragma unroll
    for (int t = 0; t < 4; ++t) acc[t] = (v8f){};

    const _Float16* arowBase = A + (size_t)(mTile + ln) * D_MODEL;

    for (int kb = 0; kb < D_MODEL; kb += 32) {
        // A fragment (16x32): lane hi half selects K sub-chunk
        const _Float16* arow = arowBase + kb + hi * 16;
        v8h a_lo = *(const v8h*)(arow);
        v8h a_hi = *(const v8h*)(arow + 8);
        v16h a   = cat16(a_lo, a_hi);
#pragma unroll
        for (int nt = 0; nt < 4; ++nt) {
            // B fragment (32x16): contiguous 32B from row n of W
            const _Float16* brow = W + (size_t)(nTile + nt * 16 + ln) * D_MODEL + kb + hi * 16;
            v16h b = *(const v16h*)brow;
            acc[nt] = wmma_f16(a, b, acc[nt]);
        }
    }

#pragma unroll
    for (int nt = 0; nt < 4; ++nt) {
#pragma unroll
        for (int r = 0; r < 8; ++r) {
            const int m = mTile + r + 8 * hi;
            const int n = nTile + nt * 16 + ln;
            const float val = acc[nt][r] + bias[n];
            if (mode == 0) {
                const int b = m >> 11, s = m & 2047, hh = n >> 6, d = n & 63;
                outH[((size_t)(b * N_HEADS + hh) * S_LEN + s) * HEAD_D + d] = (_Float16)val;
            } else if (mode == 1) {
                const int b = m >> 11, s = m & 2047, hh = n >> 6, d = n & 63;
                outH[((size_t)(b * N_HEADS + hh) * HEAD_D + d) * S_LEN + s] = (_Float16)val;
            } else {
                outF[(size_t)m * D_MODEL + n] = val;
            }
        }
    }
}

// ---------------------------------------------------------------------------
// Kernel 4: flash attention.  grid = (B*H, S/16/4), block = 128 (4 waves).
// Each wave: one 16-query block; streams keys 32 at a time with online
// softmax; accumulators stay in f32 WMMA C tiles; scores never hit HBM.
// Output: f16 [8192, 512] (token-major, col = h*64+d) for the final GEMM.
// ---------------------------------------------------------------------------
__global__ void flash_attn(const _Float16* __restrict__ Q,   // [B,H,S,Dh]
                           const _Float16* __restrict__ K,   // [B,H,S,Dh]
                           const _Float16* __restrict__ Vt,  // [B,H,Dh,S]
                           _Float16* __restrict__ attnOut) { // [8192,512]
    __shared__ __align__(16) float    Sbuf[4][16][32];
    __shared__ __align__(16) _Float16 Pbuf[4][16][32];
    __shared__ float alphaBuf[4][16];
    __shared__ float lBuf[4][16];

    const int w    = threadIdx.x >> 5;
    const int lane = threadIdx.x & 31;
    const int ln   = lane & 15;
    const int hi   = lane >> 4;

    const int bh = blockIdx.x;            // 0..31
    const int b  = bh >> 3;
    const int h  = bh & 7;
    const int m0 = (blockIdx.y * 4 + w) * 16;

    const _Float16* Qh  = Q  + (size_t)bh * S_LEN * HEAD_D;
    const _Float16* Kh  = K  + (size_t)bh * S_LEN * HEAD_D;
    const _Float16* Vth = Vt + (size_t)bh * HEAD_D * S_LEN;

    // Q fragments for the whole key loop (row = m0+ln, Dh=64 -> two K-chunks)
    const _Float16* qrow = Qh + (size_t)(m0 + ln) * HEAD_D;
    const v16h aq0 = cat16(*(const v8h*)(qrow + hi * 16),
                           *(const v8h*)(qrow + 8 + hi * 16));
    const v16h aq1 = cat16(*(const v8h*)(qrow + 32 + hi * 16),
                           *(const v8h*)(qrow + 40 + hi * 16));

    v8f o[4];
#pragma unroll
    for (int t = 0; t < 4; ++t) o[t] = (v8f){};

    float m_i = -1e30f;   // valid in lanes 0..15 (one row per lane)
    float l_i = 0.0f;
    const float scale = 0.125f;   // 1/sqrt(64)

    for (int kb = 0; kb < S_LEN; kb += 32) {
        if (kb + 32 < S_LEN)
            __builtin_prefetch(Kh + (size_t)(kb + 32 + ln) * HEAD_D, 0, 1);

        // ---- S = Q K^T for 32 keys (two 16-key N tiles) ----
#pragma unroll
        for (int nt = 0; nt < 2; ++nt) {
            const _Float16* krow = Kh + (size_t)(kb + nt * 16 + ln) * HEAD_D;
            v16h bk0 = *(const v16h*)(krow + hi * 16);
            v16h bk1 = *(const v16h*)(krow + 32 + hi * 16);
            v8f s = (v8f){};
            s = wmma_f16(aq0, bk0, s);
            s = wmma_f16(aq1, bk1, s);
#pragma unroll
            for (int r = 0; r < 8; ++r)
                Sbuf[w][r + 8 * hi][nt * 16 + ln] = s[r] * scale;
        }
        asm volatile("s_wait_dscnt 0" ::: "memory");

        // ---- online softmax: lanes 0..15 each own one query row ----
        if (lane < 16) {
            const int row = lane;
            float mx = m_i;
#pragma unroll
            for (int j = 0; j < 32; ++j) {
                float sv = Sbuf[w][row][j];
                mx = sv > mx ? sv : mx;
            }
            const float alpha = __expf(m_i - mx);
            float sum = 0.0f;
#pragma unroll
            for (int j = 0; j < 32; ++j) {
                float p = __expf(Sbuf[w][row][j] - mx);
                Pbuf[w][row][j] = (_Float16)p;
                sum += p;
            }
            l_i = l_i * alpha + sum;
            m_i = mx;
            alphaBuf[w][row] = alpha;
        }
        asm volatile("s_wait_dscnt 0" ::: "memory");

        // ---- rescale accumulators by per-row alpha ----
#pragma unroll
        for (int r = 0; r < 8; ++r) {
            const float al = alphaBuf[w][r + 8 * hi];
#pragma unroll
            for (int t = 0; t < 4; ++t) o[t][r] *= al;
        }

        // ---- P fragment in A layout from LDS ----
        const _Float16* prow = &Pbuf[w][ln][0];
        v16h ap = cat16(*(const v8h*)(prow + hi * 16),
                        *(const v8h*)(prow + 8 + hi * 16));

        // ---- O += P (16x32) x V (32x64) ----
#pragma unroll
        for (int dt = 0; dt < 4; ++dt) {
            const _Float16* vrow = Vth + (size_t)(dt * 16 + ln) * S_LEN + kb + hi * 16;
            v16h bv = *(const v16h*)vrow;
            o[dt] = wmma_f16(ap, bv, o[dt]);
        }
    }

    // ---- epilogue: divide by l, store f16 [8192,512] ----
    if (lane < 16) lBuf[w][lane] = l_i;
    asm volatile("s_wait_dscnt 0" ::: "memory");

#pragma unroll
    for (int r = 0; r < 8; ++r) {
        const int m = m0 + r + 8 * hi;
        const float inv = 1.0f / lBuf[w][r + 8 * hi];
#pragma unroll
        for (int dt = 0; dt < 4; ++dt) {
            const int col = h * HEAD_D + dt * 16 + ln;
            attnOut[(size_t)(b * S_LEN + m) * D_MODEL + col] = (_Float16)(o[dt][r] * inv);
        }
    }
}

// ---------------------------------------------------------------------------
// Launch
// ---------------------------------------------------------------------------
extern "C" void kernel_launch(void* const* d_in, const int* in_sizes, int n_in,
                              void* d_out, int out_size, void* d_ws, size_t ws_size,
                              hipStream_t stream) {
    const float* x      = (const float*)d_in[0];
    const float* gamma  = (const float*)d_in[1];
    const float* beta   = (const float*)d_in[2];
    const float* Wq     = (const float*)d_in[3];
    const float* bq     = (const float*)d_in[4];
    const float* Wk     = (const float*)d_in[5];
    const float* bk     = (const float*)d_in[6];
    const float* Wv     = (const float*)d_in[7];
    const float* bv     = (const float*)d_in[8];
    const float* Wo     = (const float*)d_in[9];
    const float* bo     = (const float*)d_in[10];
    float* out          = (float*)d_out;

    // Carve workspace (256B-aligned slices)
    char* ws = (char*)d_ws;
    size_t off = 0;
    auto carve = [&](size_t bytes) {
        void* p = ws + off;
        off += (bytes + 255) & ~(size_t)255;
        return p;
    };
    const size_t tokBytes = (size_t)N_TOK * D_MODEL * sizeof(_Float16);   // 8 MB
    const size_t wBytes   = (size_t)D_MODEL * D_MODEL * sizeof(_Float16); // 512 KB

    _Float16* xn_h   = (_Float16*)carve(tokBytes);
    _Float16* Wq_h   = (_Float16*)carve(wBytes);
    _Float16* Wk_h   = (_Float16*)carve(wBytes);
    _Float16* Wv_h   = (_Float16*)carve(wBytes);
    _Float16* Wo_h   = (_Float16*)carve(wBytes);
    _Float16* Qh     = (_Float16*)carve(tokBytes);
    _Float16* Kh     = (_Float16*)carve(tokBytes);
    _Float16* Vth    = (_Float16*)carve(tokBytes);
    _Float16* attn_h = (_Float16*)carve(tokBytes);

    // 1) weights -> f16
    cvt_weights<<<dim3(1024), dim3(256), 0, stream>>>(Wq, Wk, Wv, Wo,
                                                      Wq_h, Wk_h, Wv_h, Wo_h);
    // 2) layernorm -> f16
    layernorm_f16<<<dim3(N_TOK), dim3(256), 0, stream>>>(x, gamma, beta, xn_h);

    // 3) Q/K/V projections
    dim3 ggrid(N_TOK / 32, D_MODEL / 256);
    gemm512<<<ggrid, dim3(256), 0, stream>>>(xn_h, Wq_h, bq, 0, Qh,  nullptr);
    gemm512<<<ggrid, dim3(256), 0, stream>>>(xn_h, Wk_h, bk, 0, Kh,  nullptr);
    gemm512<<<ggrid, dim3(256), 0, stream>>>(xn_h, Wv_h, bv, 1, Vth, nullptr);

    // 4) fused flash attention
    flash_attn<<<dim3(4 * N_HEADS, S_LEN / 64), dim3(128), 0, stream>>>(Qh, Kh, Vth, attn_h);

    // 5) output projection (fp32 + bias into d_out)
    gemm512<<<ggrid, dim3(256), 0, stream>>>(attn_h, Wo_h, bo, 2, nullptr, out);
}
